// LayoutGNN_54520314855549
// MI455X (gfx1250) — matrix operand order repaired and checked by reading, not compiled, not run on hardware
//
#include <hip/hip_runtime.h>
#include <hip/hip_bf16.h>
#include <math.h>

#define NN    30000
#define MPAD  30080          // NN rounded up to 128
#define EE    480000
#define HIDC  512
#define LSTR  40             // padded LDS stride (elements): 80B rows, 16B aligned

typedef __attribute__((ext_vector_type(16))) __bf16 v16bf;
typedef __attribute__((ext_vector_type(4)))  __bf16 v4bf;
typedef __attribute__((ext_vector_type(8)))  float  v8f;

// ------------------- CDNA5 async copy: global -> LDS, 16B per lane -------------
__device__ inline void async_copy16(unsigned lds_off, const void* gptr) {
  asm volatile("global_load_async_to_lds_b128 %0, %1, off"
               :: "v"(lds_off), "v"((unsigned long long)(uintptr_t)gptr)
               : "memory");
}
__device__ inline void wait_async0() {
  asm volatile("s_wait_asynccnt 0x0" ::: "memory");
}

// ---------------- WMMA fragment loads (ISA 7.12.2 layouts, wave32) -------------

// A fragment: 16x32 bf16, LDS row-major [m][k], stride LSTR.
__device__ inline v16bf load_frag_a(const __bf16* base) {
  const int lane = threadIdx.x & 31;
  const int row  = lane & 15;
  const int hi   = lane >> 4;
  const __bf16* p = base + row * LSTR;
  v16bf r;
#pragma unroll
  for (int j = 0; j < 8; ++j) {
    int k = ((j < 4) ? (2 * j) : (16 + 2 * (j - 4))) + 8 * hi;
    r[2 * j]     = p[k];
    r[2 * j + 1] = p[k + 1];
  }
  return r;
}

// B fragment: 32x16 bf16, stored transposed in LDS as [n][k], stride LSTR.
__device__ inline v16bf load_frag_b(const __bf16* base) {
  const int lane = threadIdx.x & 31;
  const int col  = lane & 15;
  const int hi   = lane >> 4;
  const __bf16* p = base + col * LSTR;
  v16bf r;
#pragma unroll
  for (int j = 0; j < 8; ++j) {
    int k = ((j < 4) ? (2 * j) : (16 + 2 * (j - 4))) + 8 * hi;
    r[2 * j]     = p[k];
    r[2 * j + 1] = p[k + 1];
  }
  return r;
}

// ----- Tiled GEMM: C[M,Nout] = A[Mpad,K](bf16) * Bt[Nout,K](bf16, transposed) --
// Block tile 128x64, 8 waves each 32x32 via 2x2 WMMA 16x16x32 bf16.
// Double-buffered LDS, tiles staged via global_load_async_to_lds_b128.
// MODE: 0 = plain, 1 = +bias, 2 = +bias+relu.  BF16OUT: C dtype.
template <int MODE, int BF16OUT>
__global__ __launch_bounds__(256) void gemm_async_wmma(
    const __bf16* __restrict__ A, int lda,
    const __bf16* __restrict__ Bt, int K,
    void* __restrict__ Cv, int ldc,
    const float* __restrict__ bias, int M) {
  __shared__ __bf16 sA[2][128 * LSTR];
  __shared__ __bf16 sB[2][64 * LSTR];

  const int t     = threadIdx.x;
  const int lane  = t & 31;
  const int wave  = t >> 5;
  const int waveM = (wave >> 1) * 32;
  const int waveN = (wave & 1) * 32;
  const int m0    = blockIdx.x * 128;
  const int n0    = blockIdx.y * 64;

  v8f acc[2][2] = {};

  // Stage one 128x32(A) + 64x32(B) bf16 tile pair with async b128 copies.
  auto stage = [&](int buf, int k0) {
    // A: 128 rows x 64B = 512 chunks of 16B; 2 per thread
#pragma unroll
    for (int i = 0; i < 2; ++i) {
      int id = t + 256 * i;
      int r = id >> 2, c4 = id & 3;
      unsigned lo = (unsigned)(uintptr_t)(&sA[buf][r * LSTR]) + c4 * 16;
      const void* g = (const char*)A + 2 * ((size_t)(m0 + r) * lda + k0) + c4 * 16;
      async_copy16(lo, g);
    }
    // B: 64 rows x 64B = 256 chunks; 1 per thread
    {
      int r = t >> 2, c4 = t & 3;
      unsigned lo = (unsigned)(uintptr_t)(&sB[buf][r * LSTR]) + c4 * 16;
      const void* g = (const char*)Bt + 2 * ((size_t)(n0 + r) * K + k0) + c4 * 16;
      async_copy16(lo, g);
    }
  };

  const int nk = K >> 5;
  stage(0, 0);
  for (int it = 0; it < nk; ++it) {
    const int buf = it & 1;
    wait_async0();          // this wave's async chunks are in LDS
    __syncthreads();        // everyone's chunks are in LDS
    if (it + 1 < nk) stage(buf ^ 1, (it + 1) * 32);

    v16bf af[2], bfv[2];
    af[0]  = load_frag_a(&sA[buf][(waveM + 0) * LSTR]);
    af[1]  = load_frag_a(&sA[buf][(waveM + 16) * LSTR]);
    bfv[0] = load_frag_b(&sB[buf][(waveN + 0) * LSTR]);
    bfv[1] = load_frag_b(&sB[buf][(waveN + 16) * LSTR]);

#pragma unroll
    for (int mi = 0; mi < 2; ++mi)
#pragma unroll
      for (int ni = 0; ni < 2; ++ni)
        acc[mi][ni] = __builtin_amdgcn_wmma_f32_16x16x32_bf16(
            false, af[mi], false, bfv[ni], (short)0, acc[mi][ni], false, false);
    __syncthreads();        // done reading buf before it is overwritten
  }

  const int col = lane & 15;
  const int hi  = lane >> 4;
#pragma unroll
  for (int mi = 0; mi < 2; ++mi)
#pragma unroll
    for (int ni = 0; ni < 2; ++ni) {
      int cg = n0 + waveN + ni * 16 + col;
      float bv = (MODE >= 1) ? bias[cg] : 0.0f;
#pragma unroll
      for (int r = 0; r < 8; ++r) {
        int rg = m0 + waveM + mi * 16 + hi * 8 + r;
        if (rg < M) {
          float v = acc[mi][ni][r] + bv;
          if (MODE == 2) v = v > 0.0f ? v : 0.0f;
          if (BF16OUT) ((__bf16*)Cv)[(size_t)rg * ldc + cg] = (__bf16)v;
          else         ((float*)Cv)[(size_t)rg * ldc + cg] = v;
        }
      }
    }
}

// ------------------------------ converters -------------------------------------

// packed f32 -> bf16 (n multiple of 4)
__global__ void cvt_bf16(const float4* __restrict__ in, __bf16* __restrict__ out, int n4) {
  int i = blockIdx.x * blockDim.x + threadIdx.x;
  if (i >= n4) return;
  float4 v = in[i];
  v4bf o;
  o[0] = (__bf16)v.x; o[1] = (__bf16)v.y; o[2] = (__bf16)v.z; o[3] = (__bf16)v.w;
  *((v4bf*)out + i) = o;
}

// W[K,Nout] f32 -> Wt[Nout,K] bf16
__global__ void cvt_w_t(const float* __restrict__ W, __bf16* __restrict__ Wt,
                        int K, int Nout) {
  int idx = blockIdx.x * blockDim.x + threadIdx.x;
  if (idx >= K * Nout) return;
  int n = idx / K, k = idx - n * K;
  Wt[idx] = (__bf16)W[(size_t)k * Nout + n];
}

// ------------------------------ helpers ---------------------------------------

__global__ void fill_f32(float* p, float v, int n) {
  int i = blockIdx.x * blockDim.x + threadIdx.x;
  if (i < n) p[i] = v;
}
__global__ void fill_u32(unsigned* p, unsigned v, int n) {
  int i = blockIdx.x * blockDim.x + threadIdx.x;
  if (i < n) p[i] = v;
}

__global__ void reduce_sum(const float* __restrict__ x, int n, float* out) {
  __shared__ float s[256];
  int t = threadIdx.x;
  float a = 0.0f;
  for (int i = blockIdx.x * 256 + t; i < n; i += gridDim.x * 256) a += x[i];
  s[t] = a;
  __syncthreads();
  for (int o = 128; o > 0; o >>= 1) { if (t < o) s[t] += s[t + o]; __syncthreads(); }
  if (t == 0) atomicAdd(out, s[0]);
}

// catb[:, 256:768] = bbox embedding gathers (bf16)
__global__ void gather_spat(const int* __restrict__ bbox,
                            const float* __restrict__ xe,
                            const float* __restrict__ ye,
                            __bf16* __restrict__ catb) {
  int idx = blockIdx.x * blockDim.x + threadIdx.x;
  if (idx >= NN * 512) return;
  int n = idx >> 9, j = idx & 511;
  int chunk = j >> 7, w = j & 127;
  int bi = bbox[n * 4 + chunk];
  const float* tab = (chunk & 1) ? ye : xe;
  catb[(size_t)n * 768 + 256 + j] = (__bf16)tab[bi * 128 + w];
}

__global__ void node_att(const float* __restrict__ hw,
                         const float* __restrict__ ws,
                         const float* __restrict__ wd,
                         float* __restrict__ asrc, float* __restrict__ adst) {
  int idx = blockIdx.x * blockDim.x + threadIdx.x;
  if (idx >= NN * 8) return;
  int n = idx >> 3, h = idx & 7;
  const float* p  = hw + (size_t)n * 512 + h * 64;
  const float* s_ = ws + h * 64;
  const float* d_ = wd + h * 64;
  float s = 0.0f, d = 0.0f;
#pragma unroll 8
  for (int c = 0; c < 64; ++c) { s += p[c] * s_[c]; d += p[c] * d_[c]; }
  asrc[idx] = s;
  adst[idx] = d;
}

__global__ void edge_scale(const float* __restrict__ We,
                           const float* __restrict__ ae, float* sh) {
  __shared__ float red[512];
  int t = threadIdx.x;
  red[t] = We[t] * ae[t];
  __syncthreads();
  for (int o = 32; o > 0; o >>= 1) {
    if ((t & 63) < o) red[t] += red[t + o];
    __syncthreads();
  }
  if ((t & 63) == 0) sh[t >> 6] = red[t];
}

__device__ inline unsigned enc_ord(float v) {
  unsigned u = __float_as_uint(v);
  return (u & 0x80000000u) ? ~u : (u | 0x80000000u);
}
__device__ inline float dec_ord(unsigned u) {
  return __uint_as_float((u & 0x80000000u) ? (u ^ 0x80000000u) : ~u);
}

__global__ void edge_alpha_max(const int* __restrict__ ei, const float* __restrict__ ea,
                               const float* __restrict__ sum_ea,
                               const float* __restrict__ asrc, const float* __restrict__ adst,
                               const float* __restrict__ sh,
                               float* __restrict__ alpha, unsigned* __restrict__ amax) {
  int idx = blockIdx.x * blockDim.x + threadIdx.x;
  if (idx >= (EE + NN) * 8) return;
  int e = idx >> 3, h = idx & 7;
  int s, d; float av;
  if (e < EE) { s = ei[e]; d = ei[EE + e]; av = ea[e]; }
  else        { s = d = e - EE;            av = sum_ea[0] * (1.0f / EE); }
  float v = asrc[s * 8 + h] + adst[d * 8 + h] + av * sh[h];
  v = (v > 0.0f) ? v : 0.2f * v;
  alpha[idx] = v;
  atomicMax(&amax[d * 8 + h], enc_ord(v));
}

__global__ void edge_exp_sum(const int* __restrict__ ei,
                             float* __restrict__ alpha,
                             const unsigned* __restrict__ amax,
                             float* __restrict__ denom) {
  int idx = blockIdx.x * blockDim.x + threadIdx.x;
  if (idx >= (EE + NN) * 8) return;
  int e = idx >> 3, h = idx & 7;
  int d = (e < EE) ? ei[EE + e] : (e - EE);
  float ex = __expf(alpha[idx] - dec_ord(amax[d * 8 + h]));
  alpha[idx] = ex;
  atomicAdd(&denom[d * 8 + h], ex);
}

__global__ void edge_scatter(const int* __restrict__ ei,
                             const float* __restrict__ alpha,
                             const float* __restrict__ denom,
                             const float* __restrict__ hw,
                             float* __restrict__ agg) {
  int idx = blockIdx.x * blockDim.x + threadIdx.x;
  if (idx >= (EE + NN) * 8) return;
  int e = idx >> 3, h = idx & 7;
  int s, d;
  if (e < EE) { s = ei[e]; d = ei[EE + e]; } else { s = d = e - EE; }
  float w = alpha[idx] / (denom[d * 8 + h] + 1e-16f);
  const float* hs = hw + (size_t)s * 512 + h * 64;
  float* od = agg + (size_t)d * 512 + h * 64;
#pragma unroll 8
  for (int c = 0; c < 64; ++c) atomicAdd(od + c, w * hs[c]);
}

__global__ __launch_bounds__(128) void post_ln(
    const float* __restrict__ agg, const float* __restrict__ b0,
    const float* __restrict__ b1, const float* __restrict__ g,
    const float* __restrict__ be, const float* __restrict__ h0,
    float* __restrict__ hout, int add_res) {
  __shared__ float rs_[128], rq_[128];
  int n = blockIdx.x, t = threadIdx.x;
  float v[4], ls = 0.0f, lq = 0.0f;
#pragma unroll
  for (int j2 = 0; j2 < 4; ++j2) {
    int j = t + 128 * j2;
    float x = agg[(size_t)n * 512 + j] + b0[j] + b1[j];
    x = fmaxf(x, 0.0f);
    v[j2] = x; ls += x; lq += x * x;
  }
  rs_[t] = ls; rq_[t] = lq;
  __syncthreads();
  for (int o = 64; o > 0; o >>= 1) {
    if (t < o) { rs_[t] += rs_[t + o]; rq_[t] += rq_[t + o]; }
    __syncthreads();
  }
  float mu  = rs_[0] * (1.0f / 512.0f);
  float var = rq_[0] * (1.0f / 512.0f) - mu * mu;
  float inv = rsqrtf(var + 1e-5f);
#pragma unroll
  for (int j2 = 0; j2 < 4; ++j2) {
    int j = t + 128 * j2;
    float o2 = (v[j2] - mu) * inv * g[j] + be[j];
    if (add_res) o2 += h0[(size_t)n * 512 + j];
    hout[(size_t)n * 512 + j] = o2;
  }
}

__global__ void out_proj(const float* __restrict__ h, const float* __restrict__ W,
                         const float* __restrict__ b, float* __restrict__ out) {
  int n = blockIdx.x * blockDim.x + threadIdx.x;
  if (n >= NN) return;
  float acc[5];
#pragma unroll
  for (int j = 0; j < 5; ++j) acc[j] = b[j];
  const float* p = h + (size_t)n * 512;
  for (int k = 0; k < 512; ++k) {
    float hv = p[k];
#pragma unroll
    for (int j = 0; j < 5; ++j) acc[j] += hv * W[k * 5 + j];
  }
#pragma unroll
  for (int j = 0; j < 5; ++j) out[n * 5 + j] = acc[j];
}

// ------------------------------- driver ----------------------------------------

extern "C" void kernel_launch(void* const* d_in, const int* in_sizes, int n_in,
                              void* d_out, int out_size, void* d_ws, size_t ws_size,
                              hipStream_t stream) {
  const float* x       = (const float*)d_in[0];
  const int*   bbox    = (const int*)d_in[1];
  const int*   ei_sp   = (const int*)d_in[2];
  const float* ea_sp   = (const float*)d_in[3];
  const int*   ei_di   = (const int*)d_in[4];
  const float* ea_di   = (const float*)d_in[5];
  const float* proj_W  = (const float*)d_in[6];
  const float* proj_b  = (const float*)d_in[7];
  const float* x_emb   = (const float*)d_in[8];
  const float* y_emb   = (const float*)d_in[9];
  const float* cat_W   = (const float*)d_in[10];
  const float* cat_b   = (const float*)d_in[11];
  const float* gat_W   = (const float*)d_in[12];
  const float* att_src = (const float*)d_in[13];
  const float* att_dst = (const float*)d_in[14];
  const float* edge_W  = (const float*)d_in[15];
  const float* att_edg = (const float*)d_in[16];
  const float* gat_b   = (const float*)d_in[17];
  const float* ln_g    = (const float*)d_in[18];
  const float* ln_b    = (const float*)d_in[19];
  const float* out_W   = (const float*)d_in[20];
  const float* out_b   = (const float*)d_in[21];

  // ---- workspace carve (16B aligned) ----
  char* base = (char*)d_ws;
  auto alloc_f = [&](size_t n) {
    base = (char*)(((uintptr_t)base + 15) & ~(uintptr_t)15);
    float* p = (float*)base; base += n * sizeof(float); return p;
  };
  auto alloc_b = [&](size_t n) {
    base = (char*)(((uintptr_t)base + 15) & ~(uintptr_t)15);
    __bf16* p = (__bf16*)base; base += n * sizeof(__bf16); return p;
  };
  float* h0    = alloc_f((size_t)NN * 512);
  float* hbuf  = alloc_f((size_t)NN * 512);
  float* hw    = alloc_f((size_t)NN * 512);
  float* agg   = alloc_f((size_t)NN * 512);
  float* asrc  = alloc_f((size_t)NN * 8);
  float* adst  = alloc_f((size_t)NN * 8);
  float* amax  = alloc_f((size_t)NN * 8);
  float* den   = alloc_f((size_t)NN * 8);
  float* alpha = alloc_f((size_t)(EE + NN) * 8);
  float* means = alloc_f(2);
  float* sh    = alloc_f(8);
  __bf16* xb   = alloc_b((size_t)MPAD * 768);
  __bf16* catb = alloc_b((size_t)MPAD * 768);
  __bf16* hb   = alloc_b((size_t)MPAD * 512);
  __bf16* pWt  = alloc_b((size_t)256 * 768);
  __bf16* cWt  = alloc_b((size_t)512 * 768);
  __bf16* gWt  = alloc_b((size_t)4 * 512 * 512);

  // ---- one-shot conversions ----
  cvt_bf16<<<((NN * 768 / 4) + 255) / 256, 256, 0, stream>>>((const float4*)x, xb, NN * 768 / 4);
  cvt_w_t<<<(768 * 256 + 255) / 256, 256, 0, stream>>>(proj_W, pWt, 768, 256);
  cvt_w_t<<<(768 * 512 + 255) / 256, 256, 0, stream>>>(cat_W, cWt, 768, 512);
  for (int pi = 0; pi < 4; ++pi)
    cvt_w_t<<<(512 * 512 + 255) / 256, 256, 0, stream>>>(
        gat_W + (size_t)pi * 512 * 512, gWt + (size_t)pi * 512 * 512, 512, 512);

  // ---- edge-attr sums ----
  fill_f32<<<1, 64, 0, stream>>>(means, 0.0f, 2);
  reduce_sum<<<512, 256, 0, stream>>>(ea_sp, EE, means);
  reduce_sum<<<512, 256, 0, stream>>>(ea_di, EE, means + 1);

  const int gm = MPAD / 128;  // 235
  // text = x @ proj_W + proj_b -> catb[:, 0:256] (bf16)
  gemm_async_wmma<1, 1><<<dim3(gm, 4), 256, 0, stream>>>(xb, 768, pWt, 768,
                                                         catb, 768, proj_b, NN);
  gather_spat<<<(NN * 512 + 255) / 256, 256, 0, stream>>>(bbox, x_emb, y_emb, catb);
  // h0 = relu(catb @ cat_W + cat_b) (f32)
  gemm_async_wmma<2, 0><<<dim3(gm, 8), 256, 0, stream>>>(catb, 768, cWt, 768,
                                                         h0, 512, cat_b, NN);

  union { float f; unsigned u; } niv; niv.f = -3.0e38f;
  unsigned encNeg = (niv.u & 0x80000000u) ? ~niv.u : (niv.u | 0x80000000u);

  const int eb = ((EE + NN) * 8 + 255) / 256;
  const float* hcur = h0;
  for (int i = 0; i < 2; ++i) {
    cvt_bf16<<<((NN * 512 / 4) + 255) / 256, 256, 0, stream>>>(
        (const float4*)hcur, hb, NN * 512 / 4);
    fill_f32<<<(NN * 512 + 255) / 256, 256, 0, stream>>>(agg, 0.0f, NN * 512);
    for (int tt = 0; tt < 2; ++tt) {
      int pi = i * 2 + tt;
      gemm_async_wmma<0, 0><<<dim3(gm, 8), 256, 0, stream>>>(
          hb, 512, gWt + (size_t)pi * 512 * 512, 512, hw, 512, nullptr, NN);
      node_att<<<(NN * 8 + 255) / 256, 256, 0, stream>>>(
          hw, att_src + (size_t)pi * 512, att_dst + (size_t)pi * 512, asrc, adst);
      edge_scale<<<1, 512, 0, stream>>>(edge_W + (size_t)pi * 512,
                                        att_edg + (size_t)pi * 512, sh);
      fill_u32<<<(NN * 8 + 255) / 256, 256, 0, stream>>>((unsigned*)amax, encNeg, NN * 8);
      fill_f32<<<(NN * 8 + 255) / 256, 256, 0, stream>>>(den, 0.0f, NN * 8);

      const int*   ei = tt ? ei_di : ei_sp;
      const float* ea = tt ? ea_di : ea_sp;
      edge_alpha_max<<<eb, 256, 0, stream>>>(ei, ea, means + tt, asrc, adst, sh,
                                             alpha, (unsigned*)amax);
      edge_exp_sum<<<eb, 256, 0, stream>>>(ei, alpha, (const unsigned*)amax, den);
      edge_scatter<<<eb, 256, 0, stream>>>(ei, alpha, den, hw, agg);
    }
    post_ln<<<NN, 128, 0, stream>>>(agg, gat_b + (size_t)(i * 2) * 512,
                                    gat_b + (size_t)(i * 2 + 1) * 512,
                                    ln_g + (size_t)i * 512, ln_b + (size_t)i * 512,
                                    h0, hbuf, i == 1 ? 1 : 0);
    hcur = hbuf;
  }
  out_proj<<<(NN + 255) / 256, 256, 0, stream>>>(hcur, out_W, out_b, (float*)d_out);
}
